// DisenHAN_49967649521689
// MI455X (gfx1250) — matrix operand order repaired and checked by reading, not compiled, not run on hardware
//
#include <hip/hip_runtime.h>
#include <hip/hip_bf16.h>

typedef __attribute__((ext_vector_type(16))) _Float16 v16h;
typedef __attribute__((ext_vector_type(8)))  float    v8f;

#define DEV __device__ __forceinline__

// ---------------------------------------------------------------------------
// A-operand fragment for v_wmma_f32_16x16x32_f16, built from a gathered f32
// row.  16-bit A 16x32 layout (contiguous K, matching B): lanes 0-15 carry
// M=lane with K = kStep+0..15; lanes 16-31 carry M=lane-16 with
// K = kStep+16..31.
// ---------------------------------------------------------------------------
DEV v16h load_a_frag_f32(const float* __restrict__ row, int kStep, int lane) {
  const int koff = (lane >= 16) ? 16 : 0;
  const float4* q0 = reinterpret_cast<const float4*>(row + kStep + koff);
  const float4* q1 = reinterpret_cast<const float4*>(row + kStep + koff + 8);
  float4 a0 = q0[0], a1 = q0[1];
  float4 b0 = q1[0], b1 = q1[1];
  float tmp[16] = { a0.x, a0.y, a0.z, a0.w,  a1.x, a1.y, a1.z, a1.w,
                    b0.x, b0.y, b0.z, b0.w,  b1.x, b1.y, b1.z, b1.w };
  v16h a;
#pragma unroll
  for (int h = 0; h < 16; ++h) a[h] = (_Float16)tmp[h];
  return a;
}

// ---------------------------------------------------------------------------
// B-operand fragment (32x16, K x N).  Wt is the weight matrix transposed and
// converted to f16: Wt[n][k] = W[k][n].  Lanes 0-15: column n = nBase+lane,
// K = kStep+0..15; lanes 16-31: same column, K = kStep+16..31.  Each lane
// reads 16 contiguous f16 (32 B, 32 B aligned).
// ---------------------------------------------------------------------------
DEV v16h load_b_frag(const _Float16* __restrict__ Wt, int nBase, int kStep, int lane) {
  const int n = nBase + (lane & 15);
  const _Float16* p = Wt + n * 64 + kStep + ((lane >= 16) ? 16 : 0);
  return *reinterpret_cast<const v16h*>(p);
}

// ---------------------------------------------------------------------------
// Register-resident routing (T=1 => beta==1, wp==1; proof in analysis).
// WMMA D layout: lane<16 : acc[t][r] = nh[n=r   ][d=16t+col]
//                lane>=16: acc[t][r] = nh[n=r+8 ][d=16t+col],  col=lane&15.
// Facet of d is k = d/8, so the df=8 dot is a width-8 shfl_xor reduction
// (lanes of one facet share lane/8), and the 16-neighbor softmax needs one
// width-16 exchange to merge the n=0..7 / n=8..15 halves.
// ---------------------------------------------------------------------------
DEV void routing_iters(const v8f acc[4], const float cD[4], float htD[4], bool relu) {
  const float scale = 0.35355339059327378f;   // 1/sqrt(df=8)
  for (int it = 0; it < 3; ++it) {
    float e[4][8];
#pragma unroll
    for (int t = 0; t < 4; ++t)
#pragma unroll
      for (int r = 0; r < 8; ++r) {
        float v = htD[t] * acc[t][r];
        v += __shfl_xor(v, 1, 32);
        v += __shfl_xor(v, 2, 32);
        v += __shfl_xor(v, 4, 32);
        e[t][r] = v * scale;                  // full df-dot, logits
      }
    float z[4];
#pragma unroll
    for (int t = 0; t < 4; ++t) {
      float mx = e[t][0];
#pragma unroll
      for (int r = 1; r < 8; ++r) mx = fmaxf(mx, e[t][r]);
      mx = fmaxf(mx, __shfl_xor(mx, 16, 32));     // merge n halves
      float sm = 0.f, zl = 0.f;
#pragma unroll
      for (int r = 0; r < 8; ++r) {
        float ex = __expf(e[t][r] - mx);
        sm += ex;
        zl += ex * acc[t][r];
      }
      sm += __shfl_xor(sm, 16, 32);
      zl += __shfl_xor(zl, 16, 32);
      z[t] = zl / sm;                              // sum_n softmax(e)*nh
    }
#pragma unroll
    for (int t = 0; t < 4; ++t) htD[t] = cD[t] + z[t];
  }
  if (relu) {
#pragma unroll
    for (int t = 0; t < 4; ++t) htD[t] = fmaxf(htD[t], 0.f);
  }
}

// ---------------------------------------------------------------------------
// Setup: Wt[n*64+k] = f16(W[k*64+n]) for one 64x64 matrix.  grid 16 x 256.
// ---------------------------------------------------------------------------
__global__ void __launch_bounds__(256) wt_transpose_kernel(
    const float* __restrict__ src, _Float16* __restrict__ dst) {
  const int i = blockIdx.x * 256 + threadIdx.x;   // 0..4095
  const int k = i >> 6, d = i & 63;
  dst[d * 64 + k] = (_Float16)src[k * 64 + d];
}

// ---------------------------------------------------------------------------
// Layer 1: one wave per (b,s).  Gather 16 hop-2 rows, nh = rows @ W1n via
// 8x v_wmma_f32_16x16x32_f16, hid = center @ W1c via VALU (lane owns 4 dims),
// then 3 routing iterations + relu; write h1[b*16+s][64].
// ---------------------------------------------------------------------------
__global__ void __launch_bounds__(256) disenhan_layer1_kernel(
    const float* __restrict__ ctr_table, const int* __restrict__ ctr_idx,
    const float* __restrict__ nb_table,  const int* __restrict__ nb_idx,
    const _Float16* __restrict__ WtC, const _Float16* __restrict__ WtN,
    float* __restrict__ h1_out)
{
  const int gtid = blockIdx.x * 256 + threadIdx.x;
  const int wave = gtid >> 5;          // b*16+s  (0..32767)
  const int lane = gtid & 31;
  const int col  = lane & 15;

  const int nidx = nb_idx[(size_t)wave * 16 + col];
  const float* arow = nb_table + (size_t)nidx * 64;

  v8f acc[4] = { {}, {}, {}, {} };
#pragma unroll
  for (int ks = 0; ks < 64; ks += 32) {
    v16h aF = load_a_frag_f32(arow, ks, lane);
#pragma unroll
    for (int t = 0; t < 4; ++t) {
      v16h bF = load_b_frag(WtN, 16 * t, ks, lane);
      acc[t] = __builtin_amdgcn_wmma_f32_16x16x32_f16(
          false, aF, false, bF, (short)0, acc[t], false, false);
    }
  }

  // hid[d] = sum_k center[k] * W1c[k][d] at this lane's 4 dims (d = 16t+col).
  const float* crow = ctr_table + (size_t)ctr_idx[wave] * 64;
  float cD[4], htD[4];
#pragma unroll
  for (int t = 0; t < 4; ++t) {
    const _Float16* wr = WtC + (16 * t + col) * 64;   // Wt row d: contiguous K
    float h = 0.f;
    for (int k0 = 0; k0 < 64; k0 += 8) {
#pragma unroll
      for (int k = 0; k < 8; ++k) h += crow[k0 + k] * (float)wr[k0 + k];
    }
    cD[t] = h; htD[t] = h;
  }

  routing_iters(acc, cD, htD, true);

  if (lane < 16) {
    float* o = h1_out + (size_t)wave * 64;
#pragma unroll
    for (int t = 0; t < 4; ++t) o[16 * t + col] = htD[t];
  }
}

// ---------------------------------------------------------------------------
// Layer 0: one wave per root b.  nh = h1[b, 0..15] @ W0n via WMMA, center is
// the raw root embedding (no pre-encode, no relu); write final [B,64] output.
// ---------------------------------------------------------------------------
__global__ void __launch_bounds__(256) disenhan_layer0_kernel(
    const float* __restrict__ root_table, const int* __restrict__ root_idx,
    const float* __restrict__ h1_in, const _Float16* __restrict__ Wt0,
    float* __restrict__ out)
{
  const int gtid = blockIdx.x * 256 + threadIdx.x;
  const int wave = gtid >> 5;          // b (0..2047)
  const int lane = gtid & 31;
  const int col  = lane & 15;

  const float* arow = h1_in + ((size_t)wave * 16 + col) * 64;

  v8f acc[4] = { {}, {}, {}, {} };
#pragma unroll
  for (int ks = 0; ks < 64; ks += 32) {
    v16h aF = load_a_frag_f32(arow, ks, lane);
#pragma unroll
    for (int t = 0; t < 4; ++t) {
      v16h bF = load_b_frag(Wt0, 16 * t, ks, lane);
      acc[t] = __builtin_amdgcn_wmma_f32_16x16x32_f16(
          false, aF, false, bF, (short)0, acc[t], false, false);
    }
  }

  const float* crow = root_table + (size_t)root_idx[wave] * 64;
  float cD[4], htD[4];
#pragma unroll
  for (int t = 0; t < 4; ++t) { cD[t] = crow[16 * t + col]; htD[t] = cD[t]; }

  routing_iters(acc, cD, htD, false);

  if (lane < 16) {
    float* o = out + (size_t)wave * 64;
#pragma unroll
    for (int t = 0; t < 4; ++t) o[16 * t + col] = htD[t];
  }
}

// ---------------------------------------------------------------------------
// Host launcher.
// Inputs: 0 user_table, 1 item_table, 2 Wpre0_user, 3 Wpre0_item,
//         4 Wpre1_user, 5 Wpre1_item, 6 user_idx0, 7 user_idx1, 8 user_idx2,
//         9 item_idx0, 10 item_idx1, 11 item_idx2.
// Output: user_h [2048,64] then item_h [2048,64], f32.
// ws: [0,32KB) four transposed f16 W copies; [32KB, 32KB+8MB) h1 buffer,
//     reused across the two branches (stream-serialized).
// ---------------------------------------------------------------------------
extern "C" void kernel_launch(void* const* d_in, const int* in_sizes, int n_in,
                              void* d_out, int out_size, void* d_ws, size_t ws_size,
                              hipStream_t stream) {
  (void)in_sizes; (void)n_in; (void)out_size; (void)ws_size;

  const float* user_table = (const float*)d_in[0];
  const float* item_table = (const float*)d_in[1];
  const float* Wp0u = (const float*)d_in[2];
  const float* Wp0i = (const float*)d_in[3];
  const float* Wp1u = (const float*)d_in[4];
  const float* Wp1i = (const float*)d_in[5];
  const int* u_idx0 = (const int*)d_in[6];
  const int* u_idx1 = (const int*)d_in[7];
  const int* u_idx2 = (const int*)d_in[8];
  const int* i_idx0 = (const int*)d_in[9];
  const int* i_idx1 = (const int*)d_in[10];
  const int* i_idx2 = (const int*)d_in[11];

  float* out = (float*)d_out;

  _Float16* wt   = (_Float16*)d_ws;
  _Float16* Wt0u = wt + 0 * 4096;
  _Float16* Wt0i = wt + 1 * 4096;
  _Float16* Wt1u = wt + 2 * 4096;
  _Float16* Wt1i = wt + 3 * 4096;
  float* h1 = (float*)((char*)d_ws + 4 * 4096 * sizeof(_Float16));  // 32 KB in

  const int B = 2048, N0 = 16;
  dim3 blk(256);

  wt_transpose_kernel<<<16, blk, 0, stream>>>(Wp0u, Wt0u);
  wt_transpose_kernel<<<16, blk, 0, stream>>>(Wp0i, Wt0i);
  wt_transpose_kernel<<<16, blk, 0, stream>>>(Wp1u, Wt1u);
  wt_transpose_kernel<<<16, blk, 0, stream>>>(Wp1i, Wt1i);

  // user branch: center(l1)=item rows @ Wpre1_item, neighbors=user rows @ Wpre1_user,
  // hop0 neighbors h1 @ Wpre0_item, root = user_table[user_idx0].
  disenhan_layer1_kernel<<<(B * N0) / 8, blk, 0, stream>>>(
      item_table, u_idx1, user_table, u_idx2, Wt1i, Wt1u, h1);
  disenhan_layer0_kernel<<<B / 8, blk, 0, stream>>>(
      user_table, u_idx0, h1, Wt0i, out);

  // item branch (mirrored types), h1 buffer reused (stream-ordered).
  disenhan_layer1_kernel<<<(B * N0) / 8, blk, 0, stream>>>(
      user_table, i_idx1, item_table, i_idx2, Wt1u, Wt1i, h1);
  disenhan_layer0_kernel<<<B / 8, blk, 0, stream>>>(
      item_table, i_idx0, h1, Wt0u, out + (size_t)B * 64);
}
